// NetGIN_10617159155733
// MI455X (gfx1250) — compile-verified, matrix operand order checked by reading, not run on hardware
//
#include <hip/hip_runtime.h>
#include <hip/hip_bf16.h>

typedef __attribute__((ext_vector_type(16))) _Float16 v16h;
typedef __attribute__((ext_vector_type(8)))  _Float16 v8h;
typedef __attribute__((ext_vector_type(8)))  float    v8f;

#define N_NODES 50000
#define N_EDGES 800000
#define N_GRAPH 256
#define DIM 128
#define NFEAT 4529

// ptrtoint of an addrspace(3) pointer yields the raw LDS byte offset, which is
// what GLOBAL_LOAD_ASYNC_TO_LDS_* wants in its VDST operand.
__device__ inline unsigned lds_off(const void* p) {
    return (unsigned)(unsigned long long)(__attribute__((address_space(3))) const void*)p;
}

// ---------------------------------------------------------------------------
// Weight conversion: f32 [K,128] row-major -> f16 transposed [128,K]
// ---------------------------------------------------------------------------
__global__ void cvt_transpose(const float* __restrict__ W, _Float16* __restrict__ WT, int K) {
    int idx = blockIdx.x * 256 + threadIdx.x;
    if (idx >= K * 128) return;
    int k = idx >> 7, c = idx & 127;
    WT[(size_t)c * K + k] = (_Float16)W[idx];
}

// ---------------------------------------------------------------------------
// WMMA GEMM: C[M,ldc] = act(A[M,lda] (f32, opt. relu) @ B[K,128] (f16,
// pre-transposed as BT[128,K]) + bias).
// Block = 256 threads = 8 waves; wave computes a 16x128 strip.
// B panels (32x128 halves = 8KB) are double-buffered in LDS via
// global_load_async_to_lds_b128 + s_wait_asynccnt (ASYNCcnt protocol).
// All 8 B fragments of a panel are staged into distinct registers before the
// 8 WMMAs so the ds loads batch under one s_wait_dscnt and the WMMAs issue
// back-to-back (they are mutually independent: no D->A/B RAW hazards).
// ---------------------------------------------------------------------------
template<bool RELU_A, bool RELU_OUT>
__global__ __launch_bounds__(256)
void gemm_wmma(const float* __restrict__ A, int lda,
               const _Float16* __restrict__ BT, int K,
               const float* __restrict__ bias,
               float* __restrict__ C, int ldc, int M)
{
    __shared__ _Float16 Bs[2][32 * 128];       // 2 x 8KB B panels

    const int tid  = threadIdx.x;
    const int lane = tid & 31;
    const int wave = tid >> 5;
    const int row0 = (blockIdx.x * 4 + wave) * 16;

    const int  m      = row0 + (lane & 15);
    const int  koff   = (lane >> 4) * 8;       // ISA 16-bit A/B operand layout
    const bool mvalid = (m < M);
    const float* arow = A + (size_t)m * lda;

    // cooperative panel copy: thread t moves 32B of BT[cn][kt*32+ch .. +16)
    const int cn = tid >> 1;                   // 0..127 (n index)
    const int ch = (tid & 1) * 16;             // halves within the 32-wide panel
    const _Float16* bsrc = BT + (size_t)cn * K + ch;
    const unsigned ldsdst0 = lds_off(&Bs[0][cn * 32 + ch]);
    const unsigned ldsdst1 = lds_off(&Bs[1][cn * 32 + ch]);

    const int nk = K >> 5;

    v8f zero = {0.f,0.f,0.f,0.f,0.f,0.f,0.f,0.f};
    v8f acc[8];
#pragma unroll
    for (int nt = 0; nt < 8; ++nt) acc[nt] = zero;

    // prologue: async-copy panel 0 (IOFFSET applies to both LDS and global addr)
    {
        unsigned long long g = (unsigned long long)(size_t)bsrc;
        asm volatile("global_load_async_to_lds_b128 %0, %1, off"
                     :: "v"(ldsdst0), "v"(g) : "memory");
        asm volatile("global_load_async_to_lds_b128 %0, %1, off offset:16"
                     :: "v"(ldsdst0), "v"(g) : "memory");
    }

    for (int kt = 0; kt < nk; ++kt) {
        if (kt + 1 < nk) {
            unsigned long long g = (unsigned long long)(size_t)(bsrc + (kt + 1) * 32);
            unsigned d = ((kt + 1) & 1) ? ldsdst1 : ldsdst0;
            asm volatile("global_load_async_to_lds_b128 %0, %1, off"
                         :: "v"(d), "v"(g) : "memory");
            asm volatile("global_load_async_to_lds_b128 %0, %1, off offset:16"
                         :: "v"(d), "v"(g) : "memory");
            asm volatile("s_wait_asynccnt 0x2" ::: "memory");  // panel kt done
        } else {
            asm volatile("s_wait_asynccnt 0x0" ::: "memory");
        }
        __syncthreads();                        // panel kt visible to all waves

        // A fragment: f32 row -> f16, optional fused relu
        v16h a;
        if (mvalid) {
            const float* p = arow + kt * 32 + koff;
#pragma unroll
            for (int j = 0; j < 8; ++j) {
                float x0 = p[j], x1 = p[16 + j];
                if (RELU_A) { x0 = fmaxf(x0, 0.f); x1 = fmaxf(x1, 0.f); }
                a[j] = (_Float16)x0; a[8 + j] = (_Float16)x1;
            }
        } else {
#pragma unroll
            for (int j = 0; j < 16; ++j) a[j] = (_Float16)0.f;
        }

        // stage ALL 8 B fragments first (distinct registers -> ds loads batch
        // under a single s_wait_dscnt instead of one wait per WMMA)
        const _Float16* bufp = &Bs[kt & 1][0];
        v16h bf[8];
#pragma unroll
        for (int nt = 0; nt < 8; ++nt) {
            const int n = nt * 16 + (lane & 15);
            v8h lo = *(const v8h*)(bufp + n * 32 + koff);        // ds_load_b128
            v8h hi = *(const v8h*)(bufp + n * 32 + 16 + koff);   // ds_load_b128
            bf[nt] = __builtin_shufflevector(lo, hi,
                        0,1,2,3,4,5,6,7,8,9,10,11,12,13,14,15);
        }
#pragma unroll
        for (int nt = 0; nt < 8; ++nt) {
            acc[nt] = __builtin_amdgcn_wmma_f32_16x16x32_f16(
                false, a, false, bf[nt], (short)0, acc[nt], false, false);
        }
        __syncthreads();                        // done reading before overwrite
    }

    const int crow0 = row0 + (lane >> 4) * 8;   // C/D layout: VGPR r -> M=r / r+8
    const int ccol  = lane & 15;
#pragma unroll
    for (int nt = 0; nt < 8; ++nt) {
        int col = nt * 16 + ccol;
        float bv = bias[col];
#pragma unroll
        for (int r = 0; r < 8; ++r) {
            int row = crow0 + r;
            if (row < M) {
                float v = acc[nt][r] + bv;
                if (RELU_OUT) v = fmaxf(v, 0.f);
                C[(size_t)row * ldc + col] = v;
            }
        }
    }
}

// ---------------------------------------------------------------------------
// GIN aggregation init + edge scatter (native f32 atomics)
// ---------------------------------------------------------------------------
__global__ void init_agg1(const int* __restrict__ x, const float* __restrict__ emb,
                          const float* __restrict__ b1, const float* __restrict__ epsp,
                          float* __restrict__ agg, int n) {
    int idx = blockIdx.x * 256 + threadIdx.x;
    if (idx >= n * 32) return;
    int i = idx >> 5, lane = idx & 31;
    float e = 1.f + epsp[0];
    float4 v  = ((const float4*)(emb + (size_t)x[i] * DIM))[lane];
    float4 bv = ((const float4*)b1)[lane];
    float4 o; o.x = e*v.x + bv.x; o.y = e*v.y + bv.y; o.z = e*v.z + bv.z; o.w = e*v.w + bv.w;
    ((float4*)(agg + (size_t)i * DIM))[lane] = o;
}

__global__ void init_agg2(const float* __restrict__ h, int ldh,
                          const float* __restrict__ epsp,
                          float* __restrict__ agg, int n) {
    int idx = blockIdx.x * 256 + threadIdx.x;
    if (idx >= n * 32) return;
    int i = idx >> 5, lane = idx & 31;
    float e = 1.f + epsp[0];
    float4 v = ((const float4*)(h + (size_t)i * ldh))[lane];
    float4 o; o.x = e*v.x; o.y = e*v.y; o.z = e*v.z; o.w = e*v.w;
    ((float4*)(agg + (size_t)i * DIM))[lane] = o;
}

__global__ void edge_add_embed(const int* __restrict__ ei, const int* __restrict__ x,
                               const float* __restrict__ emb, float* __restrict__ agg, int E) {
    int tid = blockIdx.x * 256 + threadIdx.x;
    int e = tid >> 5;
    if (e >= E) return;
    int lane = tid & 31;
    int src = ei[e], dst = ei[E + e];
    float4 v = ((const float4*)(emb + (size_t)x[src] * DIM))[lane];
    float* d = agg + (size_t)dst * DIM + lane * 4;
    unsafeAtomicAdd(d + 0, v.x); unsafeAtomicAdd(d + 1, v.y);
    unsafeAtomicAdd(d + 2, v.z); unsafeAtomicAdd(d + 3, v.w);
}

__global__ void edge_add_h(const int* __restrict__ ei, const float* __restrict__ h, int ldh,
                           float* __restrict__ agg, int E) {
    int tid = blockIdx.x * 256 + threadIdx.x;
    int e = tid >> 5;
    if (e >= E) return;
    int lane = tid & 31;
    int src = ei[e], dst = ei[E + e];
    float4 v = ((const float4*)(h + (size_t)src * ldh))[lane];
    float* d = agg + (size_t)dst * DIM + lane * 4;
    unsafeAtomicAdd(d + 0, v.x); unsafeAtomicAdd(d + 1, v.y);
    unsafeAtomicAdd(d + 2, v.z); unsafeAtomicAdd(d + 3, v.w);
}

// ---------------------------------------------------------------------------
// BatchNorm (training mode, biased var) over N rows x 128 cols
// ---------------------------------------------------------------------------
__global__ void zero_f32(float* p, int n) {
    int i = blockIdx.x * 256 + threadIdx.x;
    if (i < n) p[i] = 0.f;
}

__global__ void bn_stats(const float* __restrict__ h, int n, float* __restrict__ stats) {
    int col = threadIdx.x;                   // 128 threads
    int r0 = blockIdx.x * 512;
    int r1 = min(n, r0 + 512);
    float s = 0.f, q = 0.f;
    for (int r = r0; r < r1; ++r) {
        float v = h[(size_t)r * DIM + col];
        s += v; q += v * v;
    }
    unsafeAtomicAdd(stats + col, s);
    unsafeAtomicAdd(stats + DIM + col, q);
}

__global__ void bn_apply(const float* __restrict__ h, const float* __restrict__ stats, int n,
                         const float* __restrict__ g, const float* __restrict__ b,
                         float* __restrict__ out, int ldo) {
    int idx = blockIdx.x * 256 + threadIdx.x;
    if (idx >= n * DIM) return;
    int i = idx >> 7, c = idx & 127;
    float inv_n = 1.f / (float)n;
    float m = stats[c] * inv_n;
    float v = stats[DIM + c] * inv_n - m * m;
    out[(size_t)i * ldo + c] = g[c] * (h[idx] - m) * rsqrtf(v + 1e-5f) + b[c];
}

// ---------------------------------------------------------------------------
// Graph mean-pool: batch[] is sorted -> one block per graph, binary search.
// ---------------------------------------------------------------------------
__device__ inline int lowerb(const int* a, int n, int key) {
    int lo = 0, hi = n;
    while (lo < hi) { int mid = (lo + hi) >> 1; if (a[mid] < key) lo = mid + 1; else hi = mid; }
    return lo;
}

__global__ void pool_kernel(const float* __restrict__ xcat, const int* __restrict__ batch,
                            int n, float* __restrict__ pooled) {
    __shared__ int s_lo, s_hi;
    int g = blockIdx.x;
    if (threadIdx.x == 0) { s_lo = lowerb(batch, n, g); s_hi = lowerb(batch, n, g + 1); }
    __syncthreads();
    int lo = s_lo, hi = s_hi;
    int c0 = threadIdx.x * 2;                // 256 threads x 2 cols = 512
    float s0 = 0.f, s1 = 0.f;
    for (int r = lo; r < hi; ++r) {
        const float* row = xcat + (size_t)r * 512 + c0;
        s0 += row[0]; s1 += row[1];
    }
    float cnt = (float)max(hi - lo, 1);
    pooled[(size_t)g * 512 + c0]     = s0 / cnt;
    pooled[(size_t)g * 512 + c0 + 1] = s1 / cnt;
}

__global__ void fc4_dot(const float* __restrict__ p3, const float* __restrict__ W,
                        const float* __restrict__ b, float* __restrict__ out) {
    int g = threadIdx.x;                     // 1 block x 256 threads
    float s = 0.f;
    for (int j = 0; j < DIM; ++j) s += p3[(size_t)g * DIM + j] * W[j];
    out[g] = s + b[0];
}

// ---------------------------------------------------------------------------
// Host orchestration
// ---------------------------------------------------------------------------
extern "C" void kernel_launch(void* const* d_in, const int* in_sizes, int n_in,
                              void* d_out, int out_size, void* d_ws, size_t ws_size,
                              hipStream_t stream) {
    (void)in_sizes; (void)n_in; (void)out_size; (void)ws_size;

    const int*   x      = (const int*)d_in[0];
    const int*   ei[3]  = {(const int*)d_in[1], (const int*)d_in[2], (const int*)d_in[3]};
    const int*   batch  = (const int*)d_in[4];
    const float* c1_W1  = (const float*)d_in[5];
    const float* c1_b1  = (const float*)d_in[6];
    const float* c1_W2  = (const float*)d_in[7];
    const float* c1_b2  = (const float*)d_in[8];
    const float* eps1   = (const float*)d_in[9];
    const float* c_W1   = (const float*)d_in[10];
    const float* c_b1   = (const float*)d_in[11];
    const float* c_W2   = (const float*)d_in[12];
    const float* c_b2   = (const float*)d_in[13];
    const float* epsv   = (const float*)d_in[14];
    const float* mlp_Wa = (const float*)d_in[15];
    const float* mlp_ba = (const float*)d_in[16];
    const float* mlp_Wb = (const float*)d_in[17];
    const float* mlp_bb = (const float*)d_in[18];
    const float* bn_g   = (const float*)d_in[19];
    const float* bn_b   = (const float*)d_in[20];
    const float* fc1_W  = (const float*)d_in[21];
    const float* fc1_b  = (const float*)d_in[22];
    const float* fc2_W  = (const float*)d_in[23];
    const float* fc2_b  = (const float*)d_in[24];
    const float* fc3_W  = (const float*)d_in[25];
    const float* fc3_b  = (const float*)d_in[26];
    const float* fc4_W  = (const float*)d_in[27];
    const float* fc4_b  = (const float*)d_in[28];
    float* out = (float*)d_out;

    // ---- f16 transposed weight area (halves) ----
    _Float16* wbase = (_Float16*)d_ws;
    _Float16* c1w2T = wbase + 0;            // 3 x 128*128
    _Float16* cw1T  = wbase + 49152;        // 9 x 128*128
    _Float16* cw2T  = wbase + 196608;       // 9 x 128*128
    _Float16* waT   = wbase + 344064;       // 4 x 384*128
    _Float16* wbT   = wbase + 540672;       // 4 x 128*128
    _Float16* f1T   = wbase + 606208;       // 512*128
    _Float16* f2T   = wbase + 671744;       // 128*128
    _Float16* f3T   = wbase + 688128;       // 128*128  (ends at 704512 halves)

    // ---- f32 scratch ----
    float* fbase  = (float*)((char*)d_ws + 1409024);
    float* agg    = fbase;                       // 3 x N*128
    float* outs   = fbase + 19200000;            // N*384 (concat of 3 branches)
    float* tbuf   = fbase + 38400000;            // N*128
    float* hpre   = fbase + 44800000;            // N*128
    float* xcat   = fbase + 51200000;            // N*512 (reps concat; h_l = slice l)
    float* pooled = fbase + 76800000;            // 256*512
    float* p1     = fbase + 76931072;            // 256*128
    float* p2     = fbase + 76963840;
    float* p3     = fbase + 76996608;
    float* stats  = fbase + 77029376;            // 256

    const int N = N_NODES, E = N_EDGES;
    const dim3 B256(256);

    // ---- convert all GEMM weights to f16 transposed ----
    auto cvt = [&](const float* W, _Float16* WT, int K) {
        cvt_transpose<<<dim3((K * 128 + 255) / 256), B256, 0, stream>>>(W, WT, K);
    };
    for (int k = 0; k < 3; ++k) cvt(c1_W2 + k * 16384, c1w2T + k * 16384, 128);
    for (int i = 0; i < 9; ++i) cvt(c_W1 + i * 16384, cw1T + i * 16384, 128);
    for (int i = 0; i < 9; ++i) cvt(c_W2 + i * 16384, cw2T + i * 16384, 128);
    for (int l = 0; l < 4; ++l) cvt(mlp_Wa + l * 49152, waT + l * 49152, 384);
    for (int l = 0; l < 4; ++l) cvt(mlp_Wb + l * 16384, wbT + l * 16384, 128);
    cvt(fc1_W, f1T, 512); cvt(fc2_W, f2T, 128); cvt(fc3_W, f3T, 128);

    const dim3 gInit((N * 32 + 255) / 256);
    const dim3 gEdge((E * 32 + 255) / 256);
    const dim3 gBN((N * DIM + 255) / 256);
    const dim3 gStats((N + 511) / 512);

#define GEMM(RA, RO, A_, lda_, BT_, K_, bias_, C_, ldc_, M_) \
    gemm_wmma<RA, RO><<<dim3(((M_) + 63) / 64), B256, 0, stream>>>(A_, lda_, BT_, K_, bias_, C_, ldc_, M_)

    auto mlp_bn = [&](int l, float* dst_slice) {
        GEMM(false, true,  outs, 384, waT + l * 49152, 384, mlp_ba + l * DIM, tbuf, DIM, N);
        GEMM(false, false, tbuf, DIM, wbT + l * 16384, DIM, mlp_bb + l * DIM, hpre, DIM, N);
        zero_f32<<<dim3(1), B256, 0, stream>>>(stats, 256);
        bn_stats<<<gStats, dim3(128), 0, stream>>>(hpre, N, stats);
        bn_apply<<<gBN, B256, 0, stream>>>(hpre, stats, N, bn_g + l * DIM, bn_b + l * DIM,
                                           dst_slice, 512);
    };

    // ---- layer 1: onehot(x)@W1 == W1[x]; agg commutes with embedding ----
    for (int k = 0; k < 3; ++k) {
        float* aggk = agg + (size_t)k * N * DIM;
        init_agg1<<<gInit, B256, 0, stream>>>(x, c1_W1 + (size_t)k * NFEAT * DIM,
                                              c1_b1 + k * DIM, eps1 + k, aggk, N);
        edge_add_embed<<<gEdge, B256, 0, stream>>>(ei[k], x, c1_W1 + (size_t)k * NFEAT * DIM,
                                                   aggk, E);
        GEMM(true, true, aggk, DIM, c1w2T + k * 16384, DIM, c1_b2 + k * DIM,
             outs + k * DIM, 384, N);
    }
    mlp_bn(0, xcat);                              // reps[0] -> xcat cols [0,128)

    // ---- layers 2-4 ----
    for (int l = 0; l < 3; ++l) {
        const float* h = xcat + l * DIM;          // previous rep, row stride 512
        for (int k = 0; k < 3; ++k) {
            float* aggk = agg + (size_t)k * N * DIM;
            int i = 3 * l + k;
            init_agg2<<<gInit, B256, 0, stream>>>(h, 512, epsv + i, aggk, N);
            edge_add_h<<<gEdge, B256, 0, stream>>>(ei[k], h, 512, aggk, E);
            GEMM(false, true, aggk, DIM, cw1T + i * 16384, DIM, c_b1 + i * DIM, tbuf, DIM, N);
            GEMM(false, true, tbuf, DIM, cw2T + i * 16384, DIM, c_b2 + i * DIM,
                 outs + k * DIM, 384, N);
        }
        mlp_bn(l + 1, xcat + (l + 1) * DIM);
    }

    // ---- pooling + FC head ----
    pool_kernel<<<dim3(N_GRAPH), B256, 0, stream>>>(xcat, batch, N, pooled);
    GEMM(false, true, pooled, 512, f1T, 512, fc1_b, p1, DIM, N_GRAPH);
    GEMM(false, true, p1, DIM, f2T, DIM, fc2_b, p2, DIM, N_GRAPH);
    GEMM(false, true, p2, DIM, f3T, DIM, fc3_b, p3, DIM, N_GRAPH);
    fc4_dot<<<dim3(1), B256, 0, stream>>>(p3, fc4_W, fc4_b, out);
#undef GEMM
}
